// JointModel_69475390980354
// MI455X (gfx1250) — matrix operand order repaired and checked by reading, not compiled, not run on hardware
//
#include <hip/hip_runtime.h>
#include <hip/hip_bf16.h>

// ---- model constants ----
#define BB   16
#define LL   128
#define EE   256
#define HH   256
#define G3H  768     // 3*H
#define TT   10
#define TDD  32
#define RR   24
#define REE  128
#define FF   545     // 2H + TD + 1
#define FP   576     // FF padded to multiple of 32

typedef __attribute__((ext_vector_type(16))) _Float16 v16h;
typedef __attribute__((ext_vector_type(8)))  _Float16 v8h;
typedef __attribute__((ext_vector_type(4)))  _Float16 h4;
typedef __attribute__((ext_vector_type(8)))  float    v8f;

__device__ __forceinline__ float logsigf(float x) {
    return fminf(x, 0.f) - log1pf(__expf(-fabsf(x)));
}

__device__ __forceinline__ void pack8(v16h& v, int base, float4 p, float4 q) {
    v[base + 0] = (_Float16)p.x; v[base + 1] = (_Float16)p.y;
    v[base + 2] = (_Float16)p.z; v[base + 3] = (_Float16)p.w;
    v[base + 4] = (_Float16)q.x; v[base + 5] = (_Float16)q.y;
    v[base + 6] = (_Float16)q.z; v[base + 7] = (_Float16)q.w;
}

// =====================================================================
// WMMA GEMM fast path (no guards): C[m, coloff+n] = A[M,K]·Bw[N,K]^T + bias
// Requires: M,N multiples of 16, K multiple of 32, lda/ldb multiples of 4.
// One wave = one 16x16 tile; f32 loads as b128, cvt to f16, v_wmma f32 acc.
// =====================================================================
__global__ void __launch_bounds__(128) k_wmma_gemm(
        const float* __restrict__ A, const float* __restrict__ Bw,
        const float* __restrict__ bias, float* __restrict__ C,
        int M, int N, int K, int lda, int ldb, int ldc, int coloff) {
    const int lane = threadIdx.x & 31;
    const int wid  = blockIdx.x * 4 + (threadIdx.x >> 5);
    const int tilesN = N >> 4;
    const int tm = wid / tilesN, tn = wid % tilesN;
    if (tm >= (M >> 4)) return;                         // wave-uniform
    const int row0 = tm << 4, col0 = tn << 4;
    const float* Arow = A  + (size_t)(row0 + (lane & 15)) * lda + ((lane >> 4) << 3);
    const float* Brow = Bw + (size_t)(col0 + (lane & 15)) * ldb + ((lane >> 4) << 3);
    const float bi = bias ? bias[col0 + (lane & 15)] : 0.f;
    v8f acc = {};
    for (int k0 = 0; k0 < K; k0 += 32) {
        float4 a0 = *(const float4*)(Arow + k0);
        float4 a1 = *(const float4*)(Arow + k0 + 4);
        float4 a2 = *(const float4*)(Arow + k0 + 16);
        float4 a3 = *(const float4*)(Arow + k0 + 20);
        float4 b0 = *(const float4*)(Brow + k0);
        float4 b1 = *(const float4*)(Brow + k0 + 4);
        float4 b2 = *(const float4*)(Brow + k0 + 16);
        float4 b3 = *(const float4*)(Brow + k0 + 20);
        v16h a, b;
        pack8(a, 0, a0, a1); pack8(a, 8, a2, a3);
        pack8(b, 0, b0, b1); pack8(b, 8, b2, b3);
        acc = __builtin_amdgcn_wmma_f32_16x16x32_f16(false, a, false, b, (short)0, acc, false, false);
    }
#pragma unroll
    for (int v = 0; v < 8; ++v) {
        int m = row0 + v + ((lane >> 4) << 3);
        int n = col0 + (lane & 15);
        C[(size_t)m * ldc + coloff + n] = acc[v] + bi;
    }
}

// =====================================================================
// Embedding gather (float4-wide): x[b,l,:] = emb_word[text[b,l]]
// =====================================================================
__global__ void k_embed(const int* __restrict__ text, const float* __restrict__ embw,
                        float* __restrict__ x, int n4) {
    for (int i = blockIdx.x * blockDim.x + threadIdx.x; i < n4; i += gridDim.x * blockDim.x) {
        int m = i >> 6, e4 = i & 63;    // 256/4 = 64 float4 per row
        ((float4*)x)[i] = ((const float4*)(embw + (size_t)text[m] * EE))[e4];
    }
}

__global__ void k_cvt_f16(const float* __restrict__ src, _Float16* __restrict__ dst, int n4) {
    for (int i = blockIdx.x * blockDim.x + threadIdx.x; i < n4; i += gridDim.x * blockDim.x) {
        float4 v = ((const float4*)src)[i];
        h4 o; o[0] = (_Float16)v.x; o[1] = (_Float16)v.y; o[2] = (_Float16)v.z; o[3] = (_Float16)v.w;
        ((h4*)dst)[i] = o;
    }
}

// zero-pad weight [N,FF] -> [N,FP]
__global__ void k_padw(const float* __restrict__ src, float* __restrict__ dst, int N) {
    for (int i = blockIdx.x * blockDim.x + threadIdx.x; i < N * FP; i += gridDim.x * blockDim.x) {
        int n = i / FP, f = i % FP;
        dst[i] = (f < FF) ? src[(size_t)n * FF + f] : 0.f;
    }
}

// =====================================================================
// Persistent GRU direction kernel. blockIdx.x = dir (0 fwd, 1 bwd).
// =====================================================================
__global__ void __launch_bounds__(256) k_gru(
        const float* __restrict__ giBase, const _Float16* __restrict__ whhBase,
        const float* __restrict__ bhhBase, const float* __restrict__ h0Base,
        float* __restrict__ y, float* __restrict__ hnBase) {
    const int dir = blockIdx.x;
    const float*    gi  = giBase + (size_t)dir * (BB * LL * G3H);
    const _Float16* whh = whhBase + (size_t)dir * (G3H * HH);
    const float*    bhh = bhhBase + dir * G3H;
    const float*    h0  = h0Base + dir * (BB * HH);
    float*          hn  = hnBase + dir * (BB * HH);
    const int coloff = dir * HH;

    __shared__ float hbuf[BB][HH];     // 16KB
    __shared__ float ghbuf[BB][G3H];   // 48KB
    const int tid = threadIdx.x;
    const int lane = tid & 31, wave = tid >> 5;

    for (int idx = tid; idx < BB * HH; idx += 256) hbuf[idx >> 8][idx & 255] = h0[idx];
    __syncthreads();

    const float* hrow = &hbuf[lane & 15][(lane >> 4) << 4];

    for (int t = 0; t < LL; ++t) {
        const int pos = dir ? (LL - 1 - t) : t;
        for (int nt = wave * 6; nt < wave * 6 + 6; ++nt) {
            const _Float16* wrow = whh + (size_t)(nt * 16 + (lane & 15)) * HH + ((lane >> 4) << 4);
            const float bh = bhh[nt * 16 + (lane & 15)];
            v8f acc = {};
            for (int k0 = 0; k0 < HH; k0 += 32) {
                float4 h0v = *(const float4*)(hrow + k0);
                float4 h1v = *(const float4*)(hrow + k0 + 4);
                float4 h2v = *(const float4*)(hrow + k0 + 8);
                float4 h3v = *(const float4*)(hrow + k0 + 12);
                v16h a;
                pack8(a, 0, h0v, h1v); pack8(a, 8, h2v, h3v);
                v16h bw = *(const v16h*)(wrow + k0);   // 32B aligned
                acc = __builtin_amdgcn_wmma_f32_16x16x32_f16(false, a, false, bw, (short)0, acc, false, false);
            }
#pragma unroll
            for (int v = 0; v < 8; ++v)
                ghbuf[v + ((lane >> 4) << 3)][nt * 16 + (lane & 15)] = acc[v] + bh;
        }
        __syncthreads();
        for (int idx = tid; idx < BB * HH; idx += 256) {
            int b = idx >> 8, j = idx & 255;
            const float* g = &gi[(size_t)(b * LL + pos) * G3H];
            float r  = 1.f / (1.f + __expf(-(g[j]      + ghbuf[b][j])));
            float z  = 1.f / (1.f + __expf(-(g[HH + j] + ghbuf[b][HH + j])));
            float nn = tanhf(g[2 * HH + j] + r * ghbuf[b][2 * HH + j]);
            float hh = (1.f - z) * nn + z * hbuf[b][j];
            hbuf[b][j] = hh;
            y[(size_t)(b * LL + pos) * (2 * HH) + coloff + j] = hh;
        }
        __syncthreads();
    }
    for (int idx = tid; idx < BB * HH; idx += 256) hn[idx] = hbuf[idx >> 8][idx & 255];
}

// =====================================================================
// hsq[b,h] = sum_g hn[g,b,h]*wlp[g] + blp
// =====================================================================
__global__ void k_hsq(const float* __restrict__ hn, const float* __restrict__ wlp,
                      const float* __restrict__ blp, float* __restrict__ hsq) {
    int i = blockIdx.x * blockDim.x + threadIdx.x;
    if (i < BB * HH) {
        float s = blp[0];
        for (int g = 0; g < 4; ++g) s += hn[g * BB * HH + i] * wlp[g];
        hsq[i] = s;
    }
}

// attention softmax over L per batch
__global__ void k_attn(const float* __restrict__ hsq, const float* __restrict__ outsq,
                       float* __restrict__ attw) {
    int b = blockIdx.x, l = threadIdx.x;
    __shared__ float red[LL];
    const float4* hq = (const float4*)(hsq + b * HH);
    const float4* oq = (const float4*)(outsq + (size_t)(b * LL + l) * HH);
    float d = 0.f;
    for (int h = 0; h < HH / 4; ++h) {
        float4 a = hq[h], c = oq[h];
        d += a.x * c.x + a.y * c.y + a.z * c.z + a.w * c.w;
    }
    red[l] = d; __syncthreads();
    for (int s = LL / 2; s > 0; s >>= 1) { if (l < s) red[l] = fmaxf(red[l], red[l + s]); __syncthreads(); }
    float mx = red[0]; __syncthreads();
    float e = __expf(d - mx);
    red[l] = e; __syncthreads();
    for (int s = LL / 2; s > 0; s >>= 1) { if (l < s) red[l] += red[l + s]; __syncthreads(); }
    attw[b * LL + l] = e / red[0];
}

// ner_score[m,t]
__global__ void k_ner(const float* __restrict__ out, const float* __restrict__ attw,
                      const float* __restrict__ wner, const float* __restrict__ bner,
                      float* __restrict__ ner) {
    int i = blockIdx.x * blockDim.x + threadIdx.x;
    if (i < BB * LL * TT) {
        int m = i / TT, t = i % TT;
        const float*  w = &wner[t * (2 * HH + 1)];
        const float4* o = (const float4*)(out + (size_t)m * 2 * HH);
        float s = bner[t];
        for (int f = 0; f < 2 * HH / 4; ++f) {
            float4 ov = o[f];
            s += ov.x * w[4 * f] + ov.y * w[4 * f + 1] + ov.z * w[4 * f + 2] + ov.w * w[4 * f + 3];
        }
        s += w[2 * HH] * attw[m];
        ner[i] = s;
    }
}

// CRF negative log-likelihood (sum over batch) -> d_out[0]
__global__ void k_crf(const float* __restrict__ em, const int* __restrict__ tags,
                      const float* __restrict__ mask, const float* __restrict__ start,
                      const float* __restrict__ trans, const float* __restrict__ endp,
                      float* __restrict__ out0) {
    __shared__ float sTr[TT * TT], sS[TT], sE[TT], red[BB];
    int tid = threadIdx.x;
    for (int i = tid; i < TT * TT; i += 32) sTr[i] = trans[i];
    if (tid < TT) { sS[tid] = start[tid]; sE[tid] = endp[tid]; }
    __syncthreads();
    float contrib = 0.f;
    if (tid < BB) {
        int b = tid;
        const int* tg = &tags[b * LL];
        const float* mk = &mask[b * LL];
        const float* e  = &em[(size_t)b * LL * TT];
        float score = sS[tg[0]] + e[tg[0]];
        for (int t = 1; t < LL; ++t)
            score += (sTr[tg[t - 1] * TT + tg[t]] + e[t * TT + tg[t]]) * mk[t];
        float msum = 0.f; for (int t = 0; t < LL; ++t) msum += mk[t];
        score += sE[tg[(int)msum - 1]];
        float alpha[TT];
        for (int c = 0; c < TT; ++c) alpha[c] = sS[c] + e[c];
        for (int t = 1; t < LL; ++t) {
            float na[TT];
            for (int c = 0; c < TT; ++c) {
                float mx = -1e30f;
                for (int p = 0; p < TT; ++p) mx = fmaxf(mx, alpha[p] + sTr[p * TT + c]);
                float s = 0.f;
                for (int p = 0; p < TT; ++p) s += __expf(alpha[p] + sTr[p * TT + c] - mx);
                na[c] = e[t * TT + c] + mx + logf(s);
            }
            float m = mk[t];
            for (int c = 0; c < TT; ++c) alpha[c] = (m > 0.f) ? na[c] : alpha[c];
        }
        float mx = -1e30f;
        for (int c = 0; c < TT; ++c) mx = fmaxf(mx, alpha[c] + sE[c]);
        float s = 0.f;
        for (int c = 0; c < TT; ++c) s += __expf(alpha[c] + sE[c] - mx);
        contrib = score - (mx + logf(s));
    }
    if (tid < BB) red[tid] = contrib;
    __syncthreads();
    if (tid == 0) { float s = 0.f; for (int i = 0; i < BB; ++i) s += red[i]; out0[0] = -s; }
}

// rel_in[m, f] padded to FP columns
__global__ void k_relin(const float* __restrict__ out, const int* __restrict__ tags,
                        const float* __restrict__ embtok, const float* __restrict__ attw,
                        float* __restrict__ relin, int n) {
    for (int i = blockIdx.x * blockDim.x + threadIdx.x; i < n; i += gridDim.x * blockDim.x) {
        int m = i / FP, f = i % FP;
        float v;
        if (f < 2 * HH)            v = out[(size_t)m * 2 * HH + f];
        else if (f < 2 * HH + TDD) v = embtok[(size_t)tags[m] * TDD + (f - 2 * HH)];
        else if (f == FF - 1)      v = attw[m];
        else                       v = 0.f;
        relin[i] = v;
    }
}

// =====================================================================
// Fused pairwise loss per block (b,i): tanh(A_j + C_i) -> f16 LDS -> WMMA
// vs emb_rel^T -> weighted BCE reduce. No [B,L,L,*] materialization.
// =====================================================================
__global__ void __launch_bounds__(32) k_pair(
        const float* __restrict__ Abuf, const float* __restrict__ Cbuf,
        const float* __restrict__ embrel, const int* __restrict__ gold,
        const float* __restrict__ mask, float* __restrict__ acc) {
    const int b = blockIdx.x >> 7;
    const int i = blockIdx.x & (LL - 1);
    const int lane = threadIdx.x;
    __shared__ float    cbuf[REE];
    __shared__ _Float16 uv[16][REE];
    __shared__ float    red[32];

    for (int h = lane; h < REE; h += 32) cbuf[h] = Cbuf[(size_t)(b * LL + i) * REE + h];

    // preload B fragments of emb_rel^T (k=h, n=r), r padded 24->32 (clamped load + select)
    const int r0 = lane & 15;
    const int r1 = 16 + r0;
    const int r1c = (r1 < RR) ? r1 : (RR - 1);
    const float keep1 = (r1 < RR) ? 1.f : 0.f;
    const float* e0 = embrel + (size_t)r0 * REE + ((lane >> 4) << 3);
    const float* e1 = embrel + (size_t)r1c * REE + ((lane >> 4) << 3);
    v16h bf0[4], bf1[4];
#pragma unroll
    for (int ks = 0; ks < 4; ++ks) {
        float4 p0 = *(const float4*)(e0 + ks * 32);
        float4 p1 = *(const float4*)(e0 + ks * 32 + 4);
        float4 p2 = *(const float4*)(e0 + ks * 32 + 16);
        float4 p3 = *(const float4*)(e0 + ks * 32 + 20);
        pack8(bf0[ks], 0, p0, p1); pack8(bf0[ks], 8, p2, p3);
        float4 q0 = *(const float4*)(e1 + ks * 32);
        float4 q1 = *(const float4*)(e1 + ks * 32 + 4);
        float4 q2 = *(const float4*)(e1 + ks * 32 + 16);
        float4 q3 = *(const float4*)(e1 + ks * 32 + 20);
        q0.x *= keep1; q0.y *= keep1; q0.z *= keep1; q0.w *= keep1;
        q1.x *= keep1; q1.y *= keep1; q1.z *= keep1; q1.w *= keep1;
        q2.x *= keep1; q2.y *= keep1; q2.z *= keep1; q2.w *= keep1;
        q3.x *= keep1; q3.y *= keep1; q3.z *= keep1; q3.w *= keep1;
        pack8(bf1[ks], 0, q0, q1); pack8(bf1[ks], 8, q2, q3);
    }
    const float mi = mask[b * LL + i];
    float lsum = 0.f;
    __syncthreads();
    const float4 cv = *(const float4*)&cbuf[lane * 4];

    for (int j0 = 0; j0 < LL; j0 += 16) {
        for (int jj = 0; jj < 16; ++jj) {
            float4 av = *(const float4*)(Abuf + (size_t)(b * LL + j0 + jj) * REE + lane * 4);
            h4 o;
            o[0] = (_Float16)tanhf(av.x + cv.x);
            o[1] = (_Float16)tanhf(av.y + cv.y);
            o[2] = (_Float16)tanhf(av.z + cv.z);
            o[3] = (_Float16)tanhf(av.w + cv.w);
            *(h4*)&uv[jj][lane * 4] = o;
        }
        __syncthreads();
        v8f d0 = {}, d1 = {};
#pragma unroll
        for (int ks = 0; ks < 4; ++ks) {
            v8h lo = *(const v8h*)&uv[lane & 15][ks * 32 + ((lane >> 4) << 3)];
            v8h hi = *(const v8h*)&uv[lane & 15][ks * 32 + ((lane >> 4) << 3) + 16];
            v16h af;
#pragma unroll
            for (int e = 0; e < 8; ++e) { af[e] = lo[e]; af[8 + e] = hi[e]; }
            d0 = __builtin_amdgcn_wmma_f32_16x16x32_f16(false, af, false, bf0[ks], (short)0, d0, false, false);
            d1 = __builtin_amdgcn_wmma_f32_16x16x32_f16(false, af, false, bf1[ks], (short)0, d1, false, false);
        }
#pragma unroll
        for (int v = 0; v < 8; ++v) {
            int j = j0 + v + ((lane >> 4) << 3);
            int g = gold[((size_t)(b * LL + i)) * LL + j];
            float pm = mi * mask[b * LL + j];
            {
                float x = d0[v]; int r = r0;
                float wr = r ? 50.f : 1.f, pw = r ? 20.f : 1.f;
                float go = (g == r) ? 1.f : 0.f;
                lsum += pm * (-(wr * (pw * go * logsigf(x) + (1.f - go) * logsigf(-x))));
            }
            {
                float x = d1[v]; int r = r1;
                float go = (g == r) ? 1.f : 0.f;
                float valid = (r < RR) ? 1.f : 0.f;
                lsum += valid * pm * (-(50.f * (20.f * go * logsigf(x) + (1.f - go) * logsigf(-x))));
            }
        }
        __syncthreads();
    }
    red[lane] = lsum; __syncthreads();
    if (lane == 0) {
        float s = 0.f;
        for (int q = 0; q < 32; ++q) s += red[q];
        atomicAdd(acc, s);
    }
}

__global__ void k_masksum(const float* __restrict__ mask, float* __restrict__ outp) {
    __shared__ float red[256];
    int tid = threadIdx.x;
    float s = 0.f;
    for (int i = tid; i < BB * LL; i += 256) s += mask[i];
    red[tid] = s; __syncthreads();
    for (int st = 128; st > 0; st >>= 1) { if (tid < st) red[tid] += red[tid + st]; __syncthreads(); }
    if (tid == 0) outp[0] = red[0];
}

__global__ void k_final(const float* __restrict__ acc, float* __restrict__ out) {
    out[1] = acc[0] / acc[1];
}

// =====================================================================
// Host orchestration
// =====================================================================
extern "C" void kernel_launch(void* const* d_in, const int* in_sizes, int n_in,
                              void* d_out, int out_size, void* d_ws, size_t ws_size,
                              hipStream_t stream) {
    (void)in_sizes; (void)n_in; (void)out_size; (void)ws_size;
    const int*   text    = (const int*)  d_in[0];
    const int*   tags    = (const int*)  d_in[1];
    const int*   goldrel = (const int*)  d_in[2];
    const float* mask    = (const float*)d_in[3];
    const float* h0      = (const float*)d_in[4];
    const float* embw    = (const float*)d_in[5];
    const float* embtok  = (const float*)d_in[6];
    const float* embrel  = (const float*)d_in[7];
    const float* wih0    = (const float*)d_in[8];
    const float* whh0    = (const float*)d_in[9];
    const float* bih0    = (const float*)d_in[10];
    const float* bhh0    = (const float*)d_in[11];
    const float* wih1    = (const float*)d_in[12];
    const float* whh1    = (const float*)d_in[13];
    const float* bih1    = (const float*)d_in[14];
    const float* bhh1    = (const float*)d_in[15];
    const float* whp     = (const float*)d_in[16];
    const float* bhp     = (const float*)d_in[17];
    const float* wlp     = (const float*)d_in[18];
    const float* blp     = (const float*)d_in[19];
    const float* wner    = (const float*)d_in[20];
    const float* bner    = (const float*)d_in[21];
    const float* cs      = (const float*)d_in[22];
    const float* ct      = (const float*)d_in[23];
    const float* ce      = (const float*)d_in[24];
    const float* wu      = (const float*)d_in[25];
    const float* bu      = (const float*)d_in[26];
    const float* wv      = (const float*)d_in[27];
    const float* bv      = (const float*)d_in[28];
    const float* wuv     = (const float*)d_in[29];
    const float* buv     = (const float*)d_in[30];
    float* out = (float*)d_out;

    // workspace layout (floats)
    float* ws = (float*)d_ws;
    size_t o = 0;
    float* x     = ws + o; o += (size_t)BB * LL * EE;
    float* gi0   = ws + o; o += 2ull * BB * LL * G3H;
    float* y0    = ws + o; o += (size_t)BB * LL * 2 * HH;
    float* gi1   = ws + o; o += 2ull * BB * LL * G3H;
    float* outb  = ws + o; o += (size_t)BB * LL * 2 * HH;
    float* hn    = ws + o; o += 4ull * BB * HH;
    float* outsq = ws + o; o += (size_t)BB * LL * HH;
    float* hsq   = ws + o; o += (size_t)BB * HH;
    float* attw  = ws + o; o += (size_t)BB * LL;
    float* ner   = ws + o; o += (size_t)BB * LL * TT;
    float* relin = ws + o; o += (size_t)BB * LL * FP;
    float* wupad = ws + o; o += (size_t)REE * FP;
    float* wvpad = ws + o; o += (size_t)REE * FP;
    float* U     = ws + o; o += (size_t)BB * LL * REE;
    float* V     = ws + o; o += (size_t)BB * LL * REE;
    float* Ab    = ws + o; o += (size_t)BB * LL * REE;
    float* Cb    = ws + o; o += (size_t)BB * LL * REE;
    float* accs  = ws + o; o += 2;
    _Float16* whh0h = (_Float16*)(ws + o); o += (size_t)G3H * HH;   // 2 dirs of halves
    _Float16* whh1h = (_Float16*)(ws + o); o += (size_t)G3H * HH;

    hipMemsetAsync((void*)accs, 0, 2 * sizeof(float), stream);

    const int NE4 = BB * LL * EE / 4;
    k_embed<<<(NE4 + 255) / 256, 256, 0, stream>>>(text, embw, x, NE4);
    k_cvt_f16<<<(2 * G3H * HH / 4 + 255) / 256, 256, 0, stream>>>(whh0, whh0h, 2 * G3H * HH / 4);
    k_cvt_f16<<<(2 * G3H * HH / 4 + 255) / 256, 256, 0, stream>>>(whh1, whh1h, 2 * G3H * HH / 4);
    k_padw<<<(REE * FP + 255) / 256, 256, 0, stream>>>(wu, wupad, REE);
    k_padw<<<(REE * FP + 255) / 256, 256, 0, stream>>>(wv, wvpad, REE);

    auto gemm_blocks = [](int M, int N) { return (((M >> 4) * (N >> 4)) + 3) / 4; };
    const int M = BB * LL;  // 2048

    for (int d = 0; d < 2; ++d)
        k_wmma_gemm<<<gemm_blocks(M, G3H), 128, 0, stream>>>(
            x, wih0 + (size_t)d * G3H * EE, bih0 + d * G3H,
            gi0 + (size_t)d * M * G3H, M, G3H, EE, EE, EE, G3H, 0);

    k_gru<<<2, 256, 0, stream>>>(gi0, whh0h, bhh0, h0, y0, hn);

    for (int d = 0; d < 2; ++d)
        k_wmma_gemm<<<gemm_blocks(M, G3H), 128, 0, stream>>>(
            y0, wih1 + (size_t)d * G3H * 2 * HH, bih1 + d * G3H,
            gi1 + (size_t)d * M * G3H, M, G3H, 2 * HH, 2 * HH, 2 * HH, G3H, 0);

    k_gru<<<2, 256, 0, stream>>>(gi1, whh1h, bhh1, h0 + 2 * BB * HH, outb, hn + 2 * BB * HH);

    k_wmma_gemm<<<gemm_blocks(M, HH), 128, 0, stream>>>(
        outb, whp, bhp, outsq, M, HH, 2 * HH, 2 * HH, 2 * HH, HH, 0);
    k_hsq<<<(BB * HH + 255) / 256, 256, 0, stream>>>(hn, wlp, blp, hsq);
    k_attn<<<BB, LL, 0, stream>>>(hsq, outsq, attw);

    k_ner<<<(M * TT + 255) / 256, 256, 0, stream>>>(outb, attw, wner, bner, ner);
    k_crf<<<1, 32, 0, stream>>>(ner, tags, mask, cs, ct, ce, out);

    const int NR = M * FP;
    k_relin<<<(NR + 255) / 256, 256, 0, stream>>>(outb, tags, embtok, attw, relin, NR);
    k_wmma_gemm<<<gemm_blocks(M, REE), 128, 0, stream>>>(relin, wupad, bu, U, M, REE, FP, FP, FP, REE, 0);
    k_wmma_gemm<<<gemm_blocks(M, REE), 128, 0, stream>>>(relin, wvpad, bv, V, M, REE, FP, FP, FP, REE, 0);
    k_wmma_gemm<<<gemm_blocks(M, REE), 128, 0, stream>>>(U, wuv,       nullptr, Ab, M, REE, REE, REE, 2 * REE, REE, 0);
    k_wmma_gemm<<<gemm_blocks(M, REE), 128, 0, stream>>>(V, wuv + REE, buv,     Cb, M, REE, REE, REE, 2 * REE, REE, 0);

    k_pair<<<BB * LL, 32, 0, stream>>>(Ab, Cb, embrel, goldrel, mask, accs);
    k_masksum<<<1, 256, 0, stream>>>(mask, accs + 1);
    k_final<<<1, 1, 0, stream>>>(accs, out);
}
